// _STELinear_6717328851718
// MI455X (gfx1250) — matrix operand order repaired and checked by reading, not compiled, not run on hardware
//
#include <hip/hip_runtime.h>

// ---------------------------------------------------------------------------
// STE ternary linear for MI455X (gfx1250):
//   scale = mean(|W|); Wt = clip(round(W/scale),-1,1)  (exact in f16)
//   Y = (Xh @ Wt^T) * scale + bias,  Xh = f16(X), f32 WMMA accumulators.
// GEMM: M=8192 (B*S), N=2048 (D_OUT), K=2048 (D_IN).
// Pipeline: double-buffered LDS, GLOBAL_LOAD_ASYNC_TO_LDS_B128 (ASYNCcnt)
// overlapping v_wmma_f32_16x16x32_f16 compute; one barrier per K-step.
// ---------------------------------------------------------------------------

typedef __attribute__((ext_vector_type(16))) _Float16 v16h;
typedef __attribute__((ext_vector_type(8)))  _Float16 v8h;
typedef __attribute__((ext_vector_type(4)))  _Float16 v4h;
typedef __attribute__((ext_vector_type(8)))  float    v8f;

#define M_DIM 8192
#define N_DIM 2048
#define K_DIM 2048
#define EPS   1e-10f

// block tile 128x256, K-step 32; 8 waves (2x4), wave tile 64x64 = 4x4 WMMA
#define BM 128
#define BN 256
#define BK 32
#define LDT 40   // BK + 8 halves pad: 80B row stride -> conflict-free 16B LDS loads

// ---------------------------------------------------------------------------
// 1) deterministic partial abs-sum of W (fixed reduction tree, no atomics)
// ---------------------------------------------------------------------------
__global__ __launch_bounds__(256) void abs_sum_partial(const float* __restrict__ w,
                                                       float* __restrict__ partials) {
    __shared__ float red[8];
    const size_t total4 = (size_t)N_DIM * K_DIM / 4;   // 1,048,576 float4
    float s = 0.0f;
    for (size_t i = (size_t)blockIdx.x * 256 + threadIdx.x; i < total4;
         i += (size_t)gridDim.x * 256) {
        float4 v = ((const float4*)w)[i];
        s += fabsf(v.x) + fabsf(v.y) + fabsf(v.z) + fabsf(v.w);
    }
    #pragma unroll
    for (int off = 16; off > 0; off >>= 1) s += __shfl_down(s, off, 32);
    if ((threadIdx.x & 31) == 0) red[threadIdx.x >> 5] = s;
    __syncthreads();
    if (threadIdx.x == 0) {
        float t = 0.0f;
        #pragma unroll
        for (int i = 0; i < 8; ++i) t += red[i];
        partials[blockIdx.x] = t;
    }
}

__global__ __launch_bounds__(256) void finalize_scale(const float* __restrict__ partials,
                                                      float* __restrict__ scale_out) {
    __shared__ float red[8];
    float s = 0.0f;
    for (int i = threadIdx.x; i < 1024; i += 256) s += partials[i];
    #pragma unroll
    for (int off = 16; off > 0; off >>= 1) s += __shfl_down(s, off, 32);
    if ((threadIdx.x & 31) == 0) red[threadIdx.x >> 5] = s;
    __syncthreads();
    if (threadIdx.x == 0) {
        float t = 0.0f;
        #pragma unroll
        for (int i = 0; i < 8; ++i) t += red[i];
        scale_out[0] = t / (float)((size_t)N_DIM * K_DIM);   // mean(|W|)
    }
}

// ---------------------------------------------------------------------------
// 2) ternary-quantize W into f16 (values {-1,0,1} are exact in f16)
// ---------------------------------------------------------------------------
__global__ __launch_bounds__(256) void quantize_w(const float* __restrict__ w,
                                                  const float* __restrict__ scale_p,
                                                  _Float16* __restrict__ wq) {
    const float inv = 1.0f / (scale_p[0] + EPS);
    size_t i = (size_t)blockIdx.x * 256 + threadIdx.x;   // one float4 each (1M total)
    float4 v = ((const float4*)w)[i];
    v4h q;
    q.x = (_Float16)fminf(1.0f, fmaxf(-1.0f, rintf(v.x * inv)));
    q.y = (_Float16)fminf(1.0f, fmaxf(-1.0f, rintf(v.y * inv)));
    q.z = (_Float16)fminf(1.0f, fmaxf(-1.0f, rintf(v.z * inv)));
    q.w = (_Float16)fminf(1.0f, fmaxf(-1.0f, rintf(v.w * inv)));
    ((v4h*)wq)[i] = q;
}

// ---------------------------------------------------------------------------
// 3) convert X to f16 once (reused from 192MB L2 by all 8 N-block strips)
// ---------------------------------------------------------------------------
__global__ __launch_bounds__(256) void cvt_x(const float* __restrict__ x,
                                             _Float16* __restrict__ xh) {
    size_t i = (size_t)blockIdx.x * 256 + threadIdx.x;   // one float4 each (4M total)
    float4 v = ((const float4*)x)[i];
    v4h h;
    h.x = (_Float16)v.x; h.y = (_Float16)v.y; h.z = (_Float16)v.z; h.w = (_Float16)v.w;
    ((v4h*)xh)[i] = h;
}

// ---------------------------------------------------------------------------
// async copy helpers (CDNA5 ASYNCcnt path)
// ---------------------------------------------------------------------------
__device__ __forceinline__ void async_b128(unsigned lds_byte_off,
                                           const void* gaddr) {
    // GLOBAL_LOAD_ASYNC_TO_LDS_B128: copies 16 B from global to LDS byte offset VDST, tracked by ASYNCcnt
    asm volatile("global_load_async_to_lds_b128 %0, %1, off"
                 :: "v"(lds_byte_off), "v"((unsigned long long)gaddr)
                 : "memory");
}

__device__ __forceinline__ void wait_async0() {
#if __has_builtin(__builtin_amdgcn_s_wait_asynccnt)
    __builtin_amdgcn_s_wait_asynccnt(0);
#else
    asm volatile("s_wait_asynccnt 0x0" ::: "memory");
#endif
}

// Stage one K-step tile: A 128x32 halves (2 chunks/thread), B 256x32 (4 chunks).
// __shared__ generic pointers: low 32 bits of the flat address == LDS byte
// offset (shared aperture lives in the high bits), which is what VDST wants.
__device__ __forceinline__ void stage_async(const _Float16* __restrict__ Xh,
                                            const _Float16* __restrict__ Wq,
                                            int m0, int n0, int k0,
                                            _Float16* bufA, _Float16* bufB,
                                            int tid) {
    #pragma unroll
    for (int i = 0; i < 2; ++i) {
        const int c = i * 256 + tid, row = c >> 2, col = (c & 3) * 8;
        async_b128((unsigned)(size_t)&bufA[row * LDT + col],
                   &Xh[(size_t)(m0 + row) * K_DIM + k0 + col]);
    }
    #pragma unroll
    for (int i = 0; i < 4; ++i) {
        const int c = i * 256 + tid, row = c >> 2, col = (c & 3) * 8;
        async_b128((unsigned)(size_t)&bufB[row * LDT + col],
                   &Wq[(size_t)(n0 + row) * K_DIM + k0 + col]);
    }
}

// ---------------------------------------------------------------------------
// 4) WMMA GEMM: Y = (Xh @ Wt^T) * scale + bias
// ---------------------------------------------------------------------------
__global__ __launch_bounds__(256) void gemm_wmma(const _Float16* __restrict__ Xh,
                                                 const _Float16* __restrict__ Wq,
                                                 const float* __restrict__ scale_p,
                                                 const float* __restrict__ bias,
                                                 float* __restrict__ out) {
    __shared__ _Float16 sA[2][BM * LDT];   // 2 x 10240 B
    __shared__ _Float16 sB[2][BN * LDT];   // 2 x 20480 B

    const int tid  = threadIdx.x;
    const int lane = tid & 31;
    const int wid  = tid >> 5;
    const int m0   = blockIdx.y * BM;
    const int n0   = blockIdx.x * BN;
    const int wm   = (wid >> 2) * 64;   // wave M offset in block tile
    const int wn   = (wid & 3)  * 64;   // wave N offset in block tile
    const int lrow = lane & 15;
    const int kbA  = (lane >> 4) << 3;  // A frag K base: 0 / 8   (ISA 16-bit A 16x32)
    const int kbB  = (lane >> 4) << 3;  // B frag K base: 0 / 8   (matches A layout)

    v8f acc[4][4];
    #pragma unroll
    for (int i = 0; i < 4; ++i)
        #pragma unroll
        for (int j = 0; j < 4; ++j) acc[i][j] = (v8f)0.0f;

    // prologue: async-stage tile 0, wait, barrier
    stage_async(Xh, Wq, m0, n0, 0, sA[0], sB[0], tid);
    wait_async0();
    __syncthreads();

    for (int k0 = 0; k0 < K_DIM; k0 += BK) {
        const int cur = (k0 >> 5) & 1;
        // overlap: async-stage next tile into the other buffer
        if (k0 + BK < K_DIM)
            stage_async(Xh, Wq, m0, n0, k0 + BK, sA[cur ^ 1], sB[cur ^ 1], tid);

        // A fragments: lane = M row; halves [kbA..kbA+7] then [kbA+16..kbA+23]
        v16h af[4];
        #pragma unroll
        for (int i = 0; i < 4; ++i) {
            const _Float16* p = &sA[cur][(wm + i * 16 + lrow) * LDT + kbA];
            v8h lo = *(const v8h*)p;
            v8h hi = *(const v8h*)(p + 16);
            af[i] = __builtin_shufflevector(lo, hi, 0, 1, 2, 3, 4, 5, 6, 7,
                                            8, 9, 10, 11, 12, 13, 14, 15);
        }
        // B fragments: lane = N col; halves [kbB..kbB+7] then [kbB+16..kbB+23]
        v16h bf[4];
        #pragma unroll
        for (int j = 0; j < 4; ++j) {
            const _Float16* p = &sB[cur][(wn + j * 16 + lrow) * LDT + kbB];
            v8h lo = *(const v8h*)p;
            v8h hi = *(const v8h*)(p + 16);
            bf[j] = __builtin_shufflevector(lo, hi, 0, 1, 2, 3, 4, 5, 6, 7,
                                            8, 9, 10, 11, 12, 13, 14, 15);
        }
        #pragma unroll
        for (int i = 0; i < 4; ++i)
            #pragma unroll
            for (int j = 0; j < 4; ++j)
                acc[i][j] = __builtin_amdgcn_wmma_f32_16x16x32_f16(
                    false, af[i], false, bf[j], (short)0, acc[i][j], false, false);

        // own async copies done, then workgroup-wide visibility via barrier
        wait_async0();
        __syncthreads();
    }

    // epilogue: C/D layout -> m = v + (lane>=16 ? 8 : 0), n = lane%16
    const float scale = scale_p[0];
    const int mhi = (lane >> 4) << 3;
    #pragma unroll
    for (int j = 0; j < 4; ++j) {
        const int n  = n0 + wn + j * 16 + lrow;
        const float bv = bias[n];
        #pragma unroll
        for (int i = 0; i < 4; ++i) {
            const int mbase = m0 + wm + i * 16 + mhi;
            #pragma unroll
            for (int v = 0; v < 8; ++v)
                out[(size_t)(mbase + v) * N_DIM + n] = acc[i][j][v] * scale + bv;
        }
    }
}

// ---------------------------------------------------------------------------
// workspace layout
// ---------------------------------------------------------------------------
static constexpr size_t OFF_SCALE = 0;
static constexpr size_t OFF_PART  = 256;
static constexpr size_t OFF_XH    = 8192;
static constexpr size_t OFF_WQ    = OFF_XH + (size_t)M_DIM * K_DIM * sizeof(_Float16);

extern "C" void kernel_launch(void* const* d_in, const int* in_sizes, int n_in,
                              void* d_out, int out_size, void* d_ws, size_t ws_size,
                              hipStream_t stream) {
    const float* x    = (const float*)d_in[0];   // [4,2048,2048] f32
    const float* w    = (const float*)d_in[1];   // [2048,2048]  f32
    const float* bias = (const float*)d_in[2];   // [2048]       f32
    float*       out  = (float*)d_out;           // [4,2048,2048] f32

    char*     ws       = (char*)d_ws;
    float*    scale    = (float*)(ws + OFF_SCALE);
    float*    partials = (float*)(ws + OFF_PART);
    _Float16* Xh       = (_Float16*)(ws + OFF_XH);
    _Float16* Wq       = (_Float16*)(ws + OFF_WQ);

    abs_sum_partial<<<1024, 256, 0, stream>>>(w, partials);
    finalize_scale<<<1, 256, 0, stream>>>(partials, scale);
    quantize_w<<<(N_DIM * K_DIM / 4) / 256, 256, 0, stream>>>(w, scale, Wq);
    cvt_x<<<(M_DIM * K_DIM / 4) / 256, 256, 0, stream>>>(x, Xh);

    dim3 grid(N_DIM / BN, M_DIM / BM);   // 8 x 64 blocks
    gemm_wmma<<<grid, 256, 0, stream>>>(Xh, Wq, scale, bias, out);
}
